// Layer_48988396978748
// MI455X (gfx1250) — matrix operand order repaired and checked by reading, not compiled, not run on hardware
//
#include <hip/hip_runtime.h>
#include <hip/hip_bf16.h>

typedef _Float16 v16h __attribute__((ext_vector_type(16)));
typedef _Float16 h8   __attribute__((ext_vector_type(8)));
typedef float    v8f  __attribute__((ext_vector_type(8)));

#define TILE  128
#define KSTEP 32
#define LSTR  40   // LDS row stride in halves: 32 + 8 pad (multiple of 8 for h8 loads)

union frag_u { v16h v; h8 h[2]; };

// ---------------------------------------------------------------------------
// Generic WMMA GEMM:  C[m,n] = alpha * sum_k A[m,k]*B(n,k)  (+bias)(+ReLU)(+res)
//   A: row-major [M,K], f16 (A16) or f32 (converted to f16 in LDS stage)
//   B: NK layout  -> source rows are N, K contiguous (weights W[n,k], K-heads)
//      KN layout  -> source rows are K, N contiguous (V matrix)   (f16 only)
//   batch offsets per blockIdx.z:  off = (z>>4)*sB + (z&15)*sH
//   Double-buffered LDS: stage tile k+1 while WMMAs consume tile k.
// ---------------------------------------------------------------------------
template<bool A16, bool B16, bool B_KN, bool HAS_BIAS, bool DO_RELU,
         bool HAS_RES, bool OUT32, bool OUT16>
__global__ __launch_bounds__(256)
void gemm_wmma(const _Float16* __restrict__ A16p, const float* __restrict__ A32p,
               const _Float16* __restrict__ B16p, const float* __restrict__ B32p,
               const float* __restrict__ bias,   const float* __restrict__ resid,
               float* __restrict__ C32, _Float16* __restrict__ C16,
               int M, int N, int K,
               long lda, long ldb, long ldc,
               long aSB, long aSH, long bSB, long bSH, long cSB, long cSH,
               float alpha)
{
    __shared__ __align__(16) _Float16 As[2][TILE * LSTR];
    __shared__ __align__(16) _Float16 Bs[2][TILE * LSTR];

    const int tid  = threadIdx.x;
    const int wave = tid >> 5;
    const int lane = tid & 31;
    const int lr   = lane & 15;
    const int hi   = lane >> 4;
    const int mw   = (wave & 3) * 32;   // wave's M offset inside 128 tile
    const int nw   = (wave >> 2) * 64;  // wave's N offset inside 128 tile

    const int  z    = blockIdx.z;
    const long aOff = (long)(z >> 4) * aSB + (long)(z & 15) * aSH;
    const long bOff = (long)(z >> 4) * bSB + (long)(z & 15) * bSH;
    const long cOff = (long)(z >> 4) * cSB + (long)(z & 15) * cSH;

    const int m0 = blockIdx.y * TILE;
    const int n0 = blockIdx.x * TILE;

    v8f acc[2][4] = {};

    // ---- stage one 128x32 A tile + B tile into LDS buffer `buf` ----
    auto stage = [&](int buf, int k0) {
        // A tile: As[buf][row][k]
        {
            const int row  = tid >> 1;
            const int half = (tid & 1) * 16;
            h8* dst = reinterpret_cast<h8*>(&As[buf][row * LSTR + half]);
            if constexpr (A16) {
                const h8* src = reinterpret_cast<const h8*>(
                    A16p + aOff + (long)(m0 + row) * lda + k0 + half);
                dst[0] = src[0];
                dst[1] = src[1];
                if (k0 + KSTEP < K)
                    __builtin_prefetch((const void*)(src + 4), 0, 1);
            } else {
                const float4* src = reinterpret_cast<const float4*>(
                    A32p + aOff + (long)(m0 + row) * lda + k0 + half);
                float4 f0 = src[0], f1 = src[1], f2 = src[2], f3 = src[3];
                h8 lo = {(_Float16)f0.x, (_Float16)f0.y, (_Float16)f0.z, (_Float16)f0.w,
                         (_Float16)f1.x, (_Float16)f1.y, (_Float16)f1.z, (_Float16)f1.w};
                h8 hv = {(_Float16)f2.x, (_Float16)f2.y, (_Float16)f2.z, (_Float16)f2.w,
                         (_Float16)f3.x, (_Float16)f3.y, (_Float16)f3.z, (_Float16)f3.w};
                dst[0] = lo;
                dst[1] = hv;
                if (k0 + KSTEP < K)
                    __builtin_prefetch((const void*)(src + 8), 0, 1);
            }
        }
        // B tile: Bs[buf][n][k]
        if constexpr (!B_KN) {
            const int row  = tid >> 1;
            const int half = (tid & 1) * 16;
            h8* dst = reinterpret_cast<h8*>(&Bs[buf][row * LSTR + half]);
            if constexpr (B16) {
                const h8* src = reinterpret_cast<const h8*>(
                    B16p + bOff + (long)(n0 + row) * ldb + k0 + half);
                dst[0] = src[0];
                dst[1] = src[1];
                if (k0 + KSTEP < K)
                    __builtin_prefetch((const void*)(src + 4), 0, 1);
            } else {
                const float4* src = reinterpret_cast<const float4*>(
                    B32p + bOff + (long)(n0 + row) * ldb + k0 + half);
                float4 f0 = src[0], f1 = src[1], f2 = src[2], f3 = src[3];
                h8 lo = {(_Float16)f0.x, (_Float16)f0.y, (_Float16)f0.z, (_Float16)f0.w,
                         (_Float16)f1.x, (_Float16)f1.y, (_Float16)f1.z, (_Float16)f1.w};
                h8 hv = {(_Float16)f2.x, (_Float16)f2.y, (_Float16)f2.z, (_Float16)f2.w,
                         (_Float16)f3.x, (_Float16)f3.y, (_Float16)f3.z, (_Float16)f3.w};
                dst[0] = lo;
                dst[1] = hv;
                if (k0 + KSTEP < K)
                    __builtin_prefetch((const void*)(src + 8), 0, 1);
            }
        } else {
            // B source is [K][N], N contiguous (V matrix): transpose into Bs[n][k]
            const int kk = tid >> 3;         // 0..31
            const int nc = (tid & 7) * 16;   // 0..112
            const h8* src = reinterpret_cast<const h8*>(
                B16p + bOff + (long)(k0 + kk) * ldb + n0 + nc);
            h8 x0 = src[0], x1 = src[1];
            #pragma unroll
            for (int j = 0; j < 8; ++j) Bs[buf][(nc + j)     * LSTR + kk] = x0[j];
            #pragma unroll
            for (int j = 0; j < 8; ++j) Bs[buf][(nc + 8 + j) * LSTR + kk] = x1[j];
        }
    };

    // ---- consume LDS buffer `buf`: load fragments (ISA 7.12.2 layout), 8 WMMAs ----
    auto compute = [&](int buf) {
        v16h aF[2], bF[4];
        #pragma unroll
        for (int mt = 0; mt < 2; ++mt) {
            const int rbase = (mw + mt * 16 + lr) * LSTR;
            frag_u u;
            u.h[0] = *reinterpret_cast<const h8*>(&As[buf][rbase + hi * 8]);       // K {0-7}/{8-15}
            u.h[1] = *reinterpret_cast<const h8*>(&As[buf][rbase + 16 + hi * 8]);  // K {16-23}/{24-31}
            aF[mt] = u.v;
        }
        #pragma unroll
        for (int nt = 0; nt < 4; ++nt) {
            const int cbase = (nw + nt * 16 + lr) * LSTR;
            frag_u u;
            u.h[0] = *reinterpret_cast<const h8*>(&Bs[buf][cbase + hi * 8]);       // K {0-7}/{8-15}
            u.h[1] = *reinterpret_cast<const h8*>(&Bs[buf][cbase + 16 + hi * 8]);  // K {16-23}/{24-31}
            bF[nt] = u.v;
        }
        #pragma unroll
        for (int mt = 0; mt < 2; ++mt)
            #pragma unroll
            for (int nt = 0; nt < 4; ++nt)
                acc[mt][nt] = __builtin_amdgcn_wmma_f32_16x16x32_f16(
                    false, aF[mt], false, bF[nt], (short)0, acc[mt][nt], false, false);
    };

    // ---- software-pipelined main loop (one barrier per K step) ----
    int cur = 0;
    stage(cur, 0);
    for (int k0 = 0; k0 < K; k0 += KSTEP) {
        __syncthreads();                       // make buf `cur` stores visible;
                                               // prior reads of `cur^1` are done
        if (k0 + KSTEP < K) stage(cur ^ 1, k0 + KSTEP);
        compute(cur);
        cur ^= 1;
    }

    // ---- epilogue: C layout (VGPR i -> M = i + 8*hi, N = lane%16) ----
    #pragma unroll
    for (int mt = 0; mt < 2; ++mt) {
        #pragma unroll
        for (int nt = 0; nt < 4; ++nt) {
            v8f c = acc[mt][nt];
            const int col = n0 + nw + nt * 16 + lr;
            float bv = 0.f;
            if constexpr (HAS_BIAS) bv = bias[col];
            #pragma unroll
            for (int i = 0; i < 8; ++i) {
                const int  row = m0 + mw + mt * 16 + hi * 8 + i;
                float v = c[i] * alpha + bv;
                if constexpr (DO_RELU) v = fmaxf(v, 0.f);
                const long idx = cOff + (long)row * ldc + col;
                if constexpr (HAS_RES) v += resid[idx];
                if constexpr (OUT32) C32[idx] = v;
                if constexpr (OUT16) C16[idx] = (_Float16)v;
            }
        }
    }
}

// ---------------------------------------------------------------------------
// LayerNorm over D=2048, one block (256 thr) per row, f16 output for WMMA A.
// ---------------------------------------------------------------------------
__global__ __launch_bounds__(256)
void ln_kernel(const float* __restrict__ x, const float* __restrict__ g,
               const float* __restrict__ b, _Float16* __restrict__ out)
{
    const int tid = threadIdx.x;
    const size_t row = blockIdx.x;
    const float* xr = x + row * 2048;

    float4 v0 = *reinterpret_cast<const float4*>(xr + tid * 8);
    float4 v1 = *reinterpret_cast<const float4*>(xr + tid * 8 + 4);
    float s  = v0.x + v0.y + v0.z + v0.w + v1.x + v1.y + v1.z + v1.w;
    float ss = v0.x*v0.x + v0.y*v0.y + v0.z*v0.z + v0.w*v0.w
             + v1.x*v1.x + v1.y*v1.y + v1.z*v1.z + v1.w*v1.w;

    __shared__ float2 red[256];
    red[tid] = make_float2(s, ss);
    __syncthreads();
    #pragma unroll
    for (int off = 128; off > 0; off >>= 1) {
        if (tid < off) {
            red[tid].x += red[tid + off].x;
            red[tid].y += red[tid + off].y;
        }
        __syncthreads();
    }
    const float mean = red[0].x * (1.f / 2048.f);
    const float var  = red[0].y * (1.f / 2048.f) - mean * mean;
    const float rs   = rsqrtf(var + 1e-5f);

    float4 g0 = *reinterpret_cast<const float4*>(g + tid * 8);
    float4 g1 = *reinterpret_cast<const float4*>(g + tid * 8 + 4);
    float4 b0 = *reinterpret_cast<const float4*>(b + tid * 8);
    float4 b1 = *reinterpret_cast<const float4*>(b + tid * 8 + 4);

    h8 o = {(_Float16)((v0.x - mean) * rs * g0.x + b0.x),
            (_Float16)((v0.y - mean) * rs * g0.y + b0.y),
            (_Float16)((v0.z - mean) * rs * g0.z + b0.z),
            (_Float16)((v0.w - mean) * rs * g0.w + b0.w),
            (_Float16)((v1.x - mean) * rs * g1.x + b1.x),
            (_Float16)((v1.y - mean) * rs * g1.y + b1.y),
            (_Float16)((v1.z - mean) * rs * g1.z + b1.z),
            (_Float16)((v1.w - mean) * rs * g1.w + b1.w)};
    *reinterpret_cast<h8*>(out + row * 2048 + tid * 8) = o;
}

// ---------------------------------------------------------------------------
// In-place row softmax over 2048 (one block per row of the score tensor).
// ---------------------------------------------------------------------------
__global__ __launch_bounds__(256)
void softmax_kernel(float* __restrict__ p)
{
    const int tid = threadIdx.x;
    float* pr = p + (size_t)blockIdx.x * 2048;

    float4 v0 = *reinterpret_cast<const float4*>(pr + tid * 8);
    float4 v1 = *reinterpret_cast<const float4*>(pr + tid * 8 + 4);

    __shared__ float red[256];
    float m = fmaxf(fmaxf(fmaxf(v0.x, v0.y), fmaxf(v0.z, v0.w)),
                    fmaxf(fmaxf(v1.x, v1.y), fmaxf(v1.z, v1.w)));
    red[tid] = m;
    __syncthreads();
    #pragma unroll
    for (int off = 128; off > 0; off >>= 1) {
        if (tid < off) red[tid] = fmaxf(red[tid], red[tid + off]);
        __syncthreads();
    }
    m = red[0];
    __syncthreads();

    v0.x = __expf(v0.x - m); v0.y = __expf(v0.y - m);
    v0.z = __expf(v0.z - m); v0.w = __expf(v0.w - m);
    v1.x = __expf(v1.x - m); v1.y = __expf(v1.y - m);
    v1.z = __expf(v1.z - m); v1.w = __expf(v1.w - m);

    red[tid] = v0.x + v0.y + v0.z + v0.w + v1.x + v1.y + v1.z + v1.w;
    __syncthreads();
    #pragma unroll
    for (int off = 128; off > 0; off >>= 1) {
        if (tid < off) red[tid] += red[tid + off];
        __syncthreads();
    }
    const float inv = 1.f / red[0];

    v0.x *= inv; v0.y *= inv; v0.z *= inv; v0.w *= inv;
    v1.x *= inv; v1.y *= inv; v1.z *= inv; v1.w *= inv;
    *reinterpret_cast<float4*>(pr + tid * 8)     = v0;
    *reinterpret_cast<float4*>(pr + tid * 8 + 4) = v1;
}

// ---------------------------------------------------------------------------
extern "C" void kernel_launch(void* const* d_in, const int* in_sizes, int n_in,
                              void* d_out, int out_size, void* d_ws, size_t ws_size,
                              hipStream_t stream)
{
    const float* x     = (const float*)d_in[0];
    const float* Wq    = (const float*)d_in[1];
    const float* bq    = (const float*)d_in[2];
    const float* Wk    = (const float*)d_in[3];
    const float* bk    = (const float*)d_in[4];
    const float* Wv    = (const float*)d_in[5];
    const float* bv    = (const float*)d_in[6];
    const float* Wo    = (const float*)d_in[7];
    const float* bo    = (const float*)d_in[8];
    const float* ln1_g = (const float*)d_in[9];
    const float* ln1_b = (const float*)d_in[10];
    const float* W1    = (const float*)d_in[11];
    const float* b1    = (const float*)d_in[12];
    const float* W2    = (const float*)d_in[13];
    const float* b2    = (const float*)d_in[14];
    const float* ln2_g = (const float*)d_in[15];
    const float* ln2_b = (const float*)d_in[16];

    const size_t R  = 4096;            // B*S rows
    const size_t D  = 2048;
    const long   S2 = 2048L * 2048L;   // per-(b,h) score slice
    const long   OS = 2048L * 128L;    // per-(b,h) O slice

    float* outX = (float*)d_out;                  // [2,2048,2048]
    float* outS = outX + (size_t)2 * 2048 * 2048; // [2,16,2048,2048]

    _Float16* xln  = (_Float16*)d_ws;
    _Float16* Qb   = xln  + R * D;
    _Float16* Kb   = Qb   + R * D;
    _Float16* Vb   = Kb   + R * D;
    _Float16* Ob   = Vb   + R * D;
    float*    x1   = (float*)(Ob + R * D);
    _Float16* xln2 = (_Float16*)(x1 + R * D);
    _Float16* hb   = xln2 + R * D;                // [4096, 8192]

    const dim3 blk(256);
    const dim3 gProj(16, 32, 1);   // N=2048, M=4096
    const dim3 gScor(16, 16, 32);  // N=2048, M=2048, z=32 heads
    const dim3 gPV(1, 16, 32);     // N=128,  M=2048, z=32
    const dim3 gFfn1(64, 32, 1);   // N=8192, M=4096
    const float iSqrtDh = 0.08838834764831845f;   // 1/sqrt(128)

    // 1) LN1 -> f16
    ln_kernel<<<4096, blk, 0, stream>>>(x, ln1_g, ln1_b, xln);

    // 2) Q/K/V = LN(x) @ W^T + b   (f16 out)
    gemm_wmma<true,false,false,true,false,false,false,true><<<gProj, blk, 0, stream>>>(
        xln, nullptr, nullptr, Wq, bq, nullptr, nullptr, Qb,
        4096, 2048, 2048, 2048, 2048, 2048, 0,0,0,0,0,0, 1.f);
    gemm_wmma<true,false,false,true,false,false,false,true><<<gProj, blk, 0, stream>>>(
        xln, nullptr, nullptr, Wk, bk, nullptr, nullptr, Kb,
        4096, 2048, 2048, 2048, 2048, 2048, 0,0,0,0,0,0, 1.f);
    gemm_wmma<true,false,false,true,false,false,false,true><<<gProj, blk, 0, stream>>>(
        xln, nullptr, nullptr, Wv, bv, nullptr, nullptr, Vb,
        4096, 2048, 2048, 2048, 2048, 2048, 0,0,0,0,0,0, 1.f);

    // 3) scores = Q K^T / sqrt(dh)  -> d_out score region (fp32), batched per (b,h)
    gemm_wmma<true,true,false,false,false,false,true,false><<<gScor, blk, 0, stream>>>(
        Qb, nullptr, Kb, nullptr, nullptr, nullptr, outS, nullptr,
        2048, 2048, 128, 2048, 2048, 2048,
        S2, 128, S2, 128, 16 * S2, S2, iSqrtDh);

    // 4) softmax in place (65536 rows)
    softmax_kernel<<<65536, blk, 0, stream>>>(outS);

    // 5) O = P @ V   (P fp32 read + converted in LDS stage; V is KN layout; O f16 [b][h][s][d])
    gemm_wmma<false,true,true,false,false,false,false,true><<<gPV, blk, 0, stream>>>(
        nullptr, outS, Vb, nullptr, nullptr, nullptr, nullptr, Ob,
        2048, 128, 2048, 2048, 2048, 128,
        16 * S2, S2, S2, 128, 16 * OS, OS, 1.f);

    // 6) x1 = x + O @ Wo^T + bo   (head-major flatten is exactly Ob viewed [4096,2048])
    gemm_wmma<true,false,false,true,false,true,true,false><<<gProj, blk, 0, stream>>>(
        Ob, nullptr, nullptr, Wo, bo, x, x1, nullptr,
        4096, 2048, 2048, 2048, 2048, 2048, 0,0,0,0,0,0, 1.f);

    // 7) LN2 -> f16
    ln_kernel<<<4096, blk, 0, stream>>>(x1, ln2_g, ln2_b, xln2);

    // 8) h = relu(LN2 @ W1^T + b1)  (f16 out, [4096,8192])
    gemm_wmma<true,false,false,true,true,false,false,true><<<gFfn1, blk, 0, stream>>>(
        xln2, nullptr, nullptr, W1, b1, nullptr, nullptr, hb,
        4096, 8192, 2048, 2048, 2048, 8192, 0,0,0,0,0,0, 1.f);

    // 9) out = x1 + h @ W2^T + b2  -> d_out x region
    gemm_wmma<true,false,false,true,false,true,true,false><<<gProj, blk, 0, stream>>>(
        hb, nullptr, nullptr, W2, b2, x1, outX, nullptr,
        4096, 2048, 8192, 8192, 8192, 2048, 0,0,0,0,0,0, 1.f);
}
